// DeformConv2D_32220844655465
// MI455X (gfx1250) — matrix operand order repaired and checked by reading, not compiled, not run on hardware
//
#include <hip/hip_runtime.h>
#include <hip/hip_bf16.h>

typedef __attribute__((ext_vector_type(16))) _Float16 v16h;
typedef __attribute__((ext_vector_type(4)))  _Float16 v4h;
typedef __attribute__((ext_vector_type(8)))  float    v8f;
typedef __attribute__((ext_vector_type(4)))  float    v4f;

#define B_  4
#define C_  32
#define H_  512
#define W_  512
#define HW_ (H_ * W_)

// x tile, channel-interleaved: xsT[(r*NCOL + col2)*32 + c]
// col2 = gx - wBase + 1: halo col2=0 (gx=wBase-1), main 1..128, halo 129.
#define NCOL 130
#define XS_ELEMS (3 * NCOL * C_)   // 12480 halves (~25 KB)
// weights, channel-interleaved per tap: wT[j9*512 + o*32 + c], rows o>=2 zero
#define WT_ELEMS (9 * 16 * C_)     // 4608 halves (~9 KB)

// One WMMA per 3x3 tap: K = channel (32). A,B fragments are contiguous
// 32B runs per lane -> 2x ds_load_b128 each; lane-half K shift lives in the
// per-lane base (no divergence).
template<int J>
__device__ __forceinline__ void gemmChunks(v8f& acc,
                                           const _Float16 (&xsT)[XS_ELEMS],
                                           const _Float16 (&wT)[WT_ELEMS],
                                           int aBase, int bBase) {
    if constexpr (J < 9) {
        constexpr int kh = J / 3;
        constexpr int kw = J - 3 * kh;
        v16h a, bm;
#pragma unroll
        for (int s = 0; s < 16; ++s) {
            const int off = s;                       // K rel: 0..15
            a[s] = wT[aBase + J * 512 + off];
        }
#pragma unroll
        for (int s = 0; s < 16; ++s)
            bm[s] = xsT[bBase + (kh * NCOL + kw) * C_ + s];
        acc = __builtin_amdgcn_wmma_f32_16x16x32_f16(
                  false, a, false, bm, (short)0, acc, false, false);
        gemmChunks<J + 1>(acc, xsT, wT, aBase, bBase);
    }
}

__global__ __launch_bounds__(256) void deform_fused_kernel(
    const float* __restrict__ x,
    const float* __restrict__ w_off,
    const float* __restrict__ b_off,
    float* __restrict__ out)
{
    __shared__ alignas(16) _Float16 xsT[XS_ELEMS];
    __shared__ alignas(16) _Float16 wT[WT_ELEMS];
    __shared__ float sOff[2 * 128];

    const int tid  = threadIdx.x;
    const int lane = tid & 31;
    const int m    = lane & 15;
    const bool lo  = (lane < 16);
    const int waveS = __builtin_amdgcn_readfirstlane(tid >> 5);  // uniform

    const int bx    = blockIdx.x;          // grid = B * H * (W/128) = 8192
    const int wTile = bx & 3;
    const int h     = (bx >> 2) & (H_ - 1);
    const int b     = bx >> 11;
    const int wBase = wTile * 128;

    // ---- stage weights (transposed): wT[j9*512 + o*32 + c] ----------------
#pragma unroll
    for (int it = 0; it < 18; ++it) {
        int i   = it * 256 + tid;           // [0, 4608)
        int c   = lane;                     // i & 31
        int gr  = it * 8 + waveS;           // i >> 5, scalar
        int o16 = gr & 15;
        int j9  = gr >> 4;
        float wv = 0.0f;
        if (o16 < 2)                        // uniform branch
            wv = w_off[o16 * 288 + c * 9 + j9];
        wT[i] = (_Float16)wv;
    }

    // ---- stage x tile with transpose: 4 channels x 4 cols per thread ------
    // item = (r*8 + g)*32 + lane : r,g scalar via waveS; 3 iterations.
#pragma unroll
    for (int it = 0; it < 3; ++it) {
        int gr = it * 8 + waveS;            // [0, 24): g = gr&7, r = gr>>3
        int g  = gr & 7;
        int r  = gr >> 3;
        int gy = h + r - 1;
        int cy = min(max(gy, 0), H_ - 1);
        v4f v0 = {}, v1 = {}, v2 = {}, v3 = {};
        if (gy == cy) {                     // uniform branch (row zero-pad)
            const float* bp = x + (((size_t)b * C_ + 4 * g) * H_ + cy) * W_
                                + wBase + 4 * lane;
            v0 = *(const v4f*)(bp);
            v1 = *(const v4f*)(bp +     (size_t)HW_);
            v2 = *(const v4f*)(bp + 2 * (size_t)HW_);
            v3 = *(const v4f*)(bp + 3 * (size_t)HW_);
        }
        int eb = (r * NCOL + 4 * lane + 1) * C_ + 4 * g;
#pragma unroll
        for (int j = 0; j < 4; ++j) {       // transpose 4x4 -> b64 stores
            v4h hv;
            hv[0] = (_Float16)v0[j];
            hv[1] = (_Float16)v1[j];
            hv[2] = (_Float16)v2[j];
            hv[3] = (_Float16)v3[j];
            *(v4h*)(&xsT[eb + j * C_]) = hv;
        }
    }
    // ---- halo columns: col2 = 0 (gx=wBase-1) and 129 (gx=wBase+128) -------
    if (tid < 192) {
        int c    = tid & 31;
        int sr   = tid >> 5;                // 0..5
        int side = sr & 1;
        int r    = sr >> 1;
        int gy = h + r - 1;
        int gx = wBase - 1 + 129 * side;
        int cy = min(max(gy, 0), H_ - 1);
        int cx = min(max(gx, 0), W_ - 1);
        float valid = ((gy == cy) & (gx == cx)) ? 1.0f : 0.0f;
        float v = x[(((size_t)b * C_ + c) * H_ + cy) * W_ + cx] * valid;
        xsT[(r * NCOL + 129 * side) * C_ + c] = (_Float16)v;
    }
    __syncthreads();

    // ---- Stage 1: offset conv = 9 taps x WMMA (K = 32 channels) -----------
    v8f acc = {};
    const int aBase = m * C_ + (lo ? 0 : 16);                    // A K-shift
    const int bBase = (waveS * 16 + m) * C_ + (lo ? 0 : 16);     // B K-shift
    gemmChunks<0>(acc, xsT, wT, aBase, bBase);

    // C: VGPR0 lanes 0-15 = (M=0,N=lane) -> dx ; VGPR1 -> dy (M=1)
    if (lo) {
        sOff[2 * (waveS * 16 + m) + 0] = acc[0] + b_off[0];
        sOff[2 * (waveS * 16 + m) + 1] = acc[1] + b_off[1];
    }
    __syncthreads();

    // ---- Stage 2: bilinear gather; p invariant, channel bases scalar ------
    const int p    = tid & 127;
    const int ch0s = __builtin_amdgcn_readfirstlane(tid >> 7);   // 0 or 1

    float ox = sOff[2 * p + 0];
    float oy = sOff[2 * p + 1];
    float ix = (float)(wBase + p) + ox;
    float iy = (float)h + oy;

    float x0f = floorf(ix), y0f = floorf(iy);
    int   x0  = (int)x0f,   y0  = (int)y0f;
    int   x1  = x0 + 1,     y1  = y0 + 1;
    float wx1 = ix - x0f, wx0 = 1.0f - wx1;
    float wy1 = iy - y0f, wy0 = 1.0f - wy1;

    float fx0 = (x0 >= 0 && x0 < W_) ? wx0 : 0.0f;
    float fx1 = (x1 >= 0 && x1 < W_) ? wx1 : 0.0f;
    float fy0 = (y0 >= 0 && y0 < H_) ? wy0 : 0.0f;
    float fy1 = (y1 >= 0 && y1 < H_) ? wy1 : 0.0f;

    int cx0 = min(max(x0, 0), W_ - 1), cx1 = min(max(x1, 0), W_ - 1);
    int cy0 = min(max(y0, 0), H_ - 1), cy1 = min(max(y1, 0), H_ - 1);

    const int o00 = cy0 * W_ + cx0, o01 = cy0 * W_ + cx1;
    const int o10 = cy1 * W_ + cx0, o11 = cy1 * W_ + cx1;
    const float w00 = fy0 * fx0, w01 = fy0 * fx1;
    const float w10 = fy1 * fx0, w11 = fy1 * fx1;

    const float* xc = x   + ((size_t)b * C_ + ch0s) * (size_t)HW_;
    float*       op = out + (((size_t)b * C_ + ch0s) * H_ + h) * W_ + wBase;

#pragma unroll
    for (int it = 0; it < 16; ++it) {       // channels ch0s, ch0s+2, ..., +30
        float r = w00 * xc[o00] + w01 * xc[o01]
                + w10 * xc[o10] + w11 * xc[o11];
        op[p] = r;
        xc += 2 * (size_t)HW_;
        op += 2 * (size_t)HW_;
    }
}

extern "C" void kernel_launch(void* const* d_in, const int* in_sizes, int n_in,
                              void* d_out, int out_size, void* d_ws, size_t ws_size,
                              hipStream_t stream) {
    (void)in_sizes; (void)n_in; (void)out_size; (void)d_ws; (void)ws_size;
    const float* x     = (const float*)d_in[0];
    const float* w_off = (const float*)d_in[1];
    const float* b_off = (const float*)d_in[2];
    float* out = (float*)d_out;

    const int blocks = B_ * H_ * (W_ / 128);   // 8192
    deform_fused_kernel<<<blocks, 256, 0, stream>>>(x, w_off, b_off, out);
}
